// LlamaDecoderLayer_52106543235533
// MI455X (gfx1250) — matrix operand order repaired and checked by reading, not compiled, not run on hardware
//
#include <hip/hip_runtime.h>
#include <hip/hip_bf16.h>
#include <cstdint>
#include <cstddef>

#define DI __device__ __forceinline__

typedef __bf16 bf16_t;
typedef __bf16 v8bf  __attribute__((ext_vector_type(8)));
typedef __bf16 v16bf __attribute__((ext_vector_type(16)));
typedef float  v8f   __attribute__((ext_vector_type(8)));
typedef unsigned int v4u __attribute__((ext_vector_type(4)));
typedef int v4i __attribute__((ext_vector_type(4)));
typedef int v8i __attribute__((ext_vector_type(8)));

#if defined(__has_builtin)
#if __has_builtin(__builtin_amdgcn_tensor_load_to_lds) && __has_builtin(__builtin_amdgcn_s_wait_tensorcnt)
#define HAVE_TDM 1
#endif
#endif
#ifndef HAVE_TDM
#define HAVE_TDM 0
#endif

namespace cfg {
constexpr int B   = 8;
constexpr int S   = 1024;
constexpr int D   = 576;
constexpr int H   = 9;
constexpr int HD  = 64;
constexpr int LAT = 144;
constexpr int LATP= 160;   // LAT padded to a multiple of 32 for WMMA K-loop
constexpr int FF  = 1536;
constexpr int NE  = 7;
constexpr int M   = B * S; // 8192 tokens
}

// ---------------------------------------------------------------------------
// WMMA fragment helpers (bf16 16x16x32, wave32)
// A (16xK): lane l<16 -> row l, K {0..7,16..23}; lane l+16 -> row l, K {8..15,24..31}
// B (Kx16): lane l<16 -> col l, K {0..7,16..23}; lane l+16 -> col l-16, K {8..15,24..31}
// C (16x16 f32): VGPR r -> row r + 8*(lane>=16); col = lane%16
// ---------------------------------------------------------------------------
DI v16bf cat8(v8bf lo, v8bf hi) {
  v16bf r;
#pragma unroll
  for (int i = 0; i < 8; ++i) { r[i] = lo[i]; r[8 + i] = hi[i]; }
  return r;
}

DI v16bf load_a(const bf16_t* A, int lda, int row, int k0, int lane) {
  const bf16_t* p = A + (size_t)row * lda + k0 + ((lane >> 4) << 3);
  return cat8(*(const v8bf*)p, *(const v8bf*)(p + 16));
}

DI v16bf load_b(const bf16_t* W, int ldb, int col, int k0, int lane) {
  const bf16_t* p = W + (size_t)col * ldb + k0 + ((lane >> 4) << 3);
  return cat8(*(const v8bf*)p, *(const v8bf*)(p + 16));
}

DI v8f wmma_bf16(v16bf a, v16bf b, v8f c) {
  return __builtin_amdgcn_wmma_f32_16x16x32_bf16(
      /*neg_a=*/false, a, /*neg_b=*/false, b,
      /*c_mod=*/(short)0, c, /*reuse_a=*/false, /*reuse_b=*/false);
}

DI v8f zero8() { v8f z = {0.f,0.f,0.f,0.f,0.f,0.f,0.f,0.f}; return z; }

// reduce across the 16 lanes of each half-wave (xor masks 1..8 stay in-half)
DI float hmax16(float x) {
#pragma unroll
  for (int m = 1; m < 16; m <<= 1) x = fmaxf(x, __shfl_xor(x, m, 32));
  return x;
}
DI float hsum16(float x) {
#pragma unroll
  for (int m = 1; m < 16; m <<= 1) x += __shfl_xor(x, m, 32);
  return x;
}

// ---------------------------------------------------------------------------
// Tensor Data Mover: async 2D panel load (rows x k_elems of bf16) into LDS.
// Descriptor per CDNA5 ISA (D# group0 128b, group1 256b).
// ---------------------------------------------------------------------------
#if HAVE_TDM
DI void tdm_load_2d(unsigned lds_byte_off, const void* gptr,
                    unsigned k_elems, unsigned rows, unsigned stride_elems) {
  unsigned long long ga = (unsigned long long)(uintptr_t)gptr;
  v4u g0;
  g0[0] = 1u;                                   // count=1 (valid user descriptor)
  g0[1] = lds_byte_off;                         // lds_addr (bytes)
  g0[2] = (unsigned)(ga & 0xFFFFFFFFull);       // global_addr[31:0]
  g0[3] = (unsigned)((ga >> 32) & 0x01FFFFFFu)  // global_addr[56:32]
          | (2u << 30);                          // type=2 ("image")
  v8i g1;
  g1[0] = (int)(1u << 16);                      // data_size=1 -> 2 bytes; no mask/flags
  g1[1] = (int)((k_elems & 0xFFFFu) << 16);     // tensor_dim0[15:0] @ bits 63:48
  g1[2] = (int)(((k_elems >> 16) & 0xFFFFu)     // tensor_dim0[31:16]
          | ((rows & 0xFFFFu) << 16));          // tensor_dim1[15:0]
  g1[3] = (int)(((rows >> 16) & 0xFFFFu)        // tensor_dim1[31:16]
          | ((k_elems & 0xFFFFu) << 16));       // tile_dim0 = full row width
  g1[4] = (int)(rows & 0xFFFFu);                // tile_dim1 = rows; tile_dim2 = 0
  g1[5] = (int)stride_elems;                    // tensor_dim0_stride[31:0]
  g1[6] = 0;                                    // stride hi / dim1_stride lo
  g1[7] = 0;
  v4i z4 = {0, 0, 0, 0};
#if __clang_major__ >= 23
  v8i z8 = {0, 0, 0, 0, 0, 0, 0, 0};
  __builtin_amdgcn_tensor_load_to_lds(g0, g1, z4, z4, z8, 0);
#else
  __builtin_amdgcn_tensor_load_to_lds(g0, g1, z4, z4, 0);
#endif
}
#endif

// ---------------------------------------------------------------------------
// f32 -> bf16 convert with optional column padding (pads with zeros)
// ---------------------------------------------------------------------------
__global__ void k_cvt(const float* __restrict__ src, bf16_t* __restrict__ dst,
                      int rows, int cols, int dld) {
  int idx = blockIdx.x * blockDim.x + threadIdx.x;
  if (idx >= rows * dld) return;
  int c = idx % dld, r = idx / dld;
  dst[idx] = (c < cols) ? (bf16_t)src[(size_t)r * cols + c] : (bf16_t)0.f;
}

// ---------------------------------------------------------------------------
// RMSNorm: f32 in, bf16 out (one block per token row)
// ---------------------------------------------------------------------------
__global__ void k_rmsnorm(const float* __restrict__ X, const float* __restrict__ w,
                          bf16_t* __restrict__ O, int Dd) {
  const int row = blockIdx.x, tid = threadIdx.x;
  __shared__ float red[256];
  const float* x = X + (size_t)row * Dd;
  float ss = 0.f;
  for (int i = tid; i < Dd; i += 256) { float v = x[i]; ss += v * v; }
  red[tid] = ss; __syncthreads();
  for (int s = 128; s > 0; s >>= 1) { if (tid < s) red[tid] += red[tid + s]; __syncthreads(); }
  float inv = rsqrtf(red[0] / (float)Dd + 1e-5f);
  bf16_t* o = O + (size_t)row * Dd;
  for (int i = tid; i < Dd; i += 256) o[i] = (bf16_t)(x[i] * inv * w[i]);
}

// ---------------------------------------------------------------------------
// 1-tile WMMA GEMM (used where N is not a multiple of 64): C = A @ W^T
// ---------------------------------------------------------------------------
__global__ void k_gemm_bf16out(const bf16_t* __restrict__ A, const bf16_t* __restrict__ W,
                               bf16_t* __restrict__ C, int lda, int ldb, int ldc, int K) {
  const int lane = threadIdx.x & 31, l = lane & 15, half = lane >> 4;
  const int m0 = blockIdx.x * 16, n0 = blockIdx.y * 16;
  v8f acc = zero8();
  for (int k0 = 0; k0 < K; k0 += 32) {
    v16bf a = load_a(A, lda, m0 + l, k0, lane);
    v16bf b = load_b(W, ldb, n0 + l, k0, lane);
    acc = wmma_bf16(a, b, acc);
  }
#pragma unroll
  for (int r = 0; r < 8; ++r)
    C[(size_t)(m0 + r + 8 * half) * ldc + n0 + l] = (bf16_t)acc[r];
}

// ---------------------------------------------------------------------------
// 4-wide WMMA GEMM: wave computes a 16x64 tile, A fragment reused 4x
// ---------------------------------------------------------------------------
__global__ void k_gemm4_bf16out(const bf16_t* __restrict__ A, const bf16_t* __restrict__ W,
                                bf16_t* __restrict__ C, int lda, int ldb, int ldc, int K) {
  const int lane = threadIdx.x & 31, l = lane & 15, half = lane >> 4;
  const int m0 = blockIdx.x * 16, n0 = blockIdx.y * 64;
  v8f acc[4] = {zero8(), zero8(), zero8(), zero8()};
  for (int k0 = 0; k0 < K; k0 += 32) {
    v16bf a = load_a(A, lda, m0 + l, k0, lane);
#pragma unroll
    for (int t = 0; t < 4; ++t)
      acc[t] = wmma_bf16(a, load_b(W, ldb, n0 + t * 16 + l, k0, lane), acc[t]);
  }
#pragma unroll
  for (int t = 0; t < 4; ++t)
#pragma unroll
    for (int r = 0; r < 8; ++r)
      C[(size_t)(m0 + r + 8 * half) * ldc + n0 + t * 16 + l] = (bf16_t)acc[t][r];
}

__global__ void k_gemm4_f32res(const bf16_t* __restrict__ A, const bf16_t* __restrict__ W,
                               const float* __restrict__ R, float* __restrict__ C,
                               int lda, int ldb, int ldc, int K) {
  const int lane = threadIdx.x & 31, l = lane & 15, half = lane >> 4;
  const int m0 = blockIdx.x * 16, n0 = blockIdx.y * 64;
  v8f acc[4] = {zero8(), zero8(), zero8(), zero8()};
  for (int k0 = 0; k0 < K; k0 += 32) {
    v16bf a = load_a(A, lda, m0 + l, k0, lane);
#pragma unroll
    for (int t = 0; t < 4; ++t)
      acc[t] = wmma_bf16(a, load_b(W, ldb, n0 + t * 16 + l, k0, lane), acc[t]);
  }
#pragma unroll
  for (int t = 0; t < 4; ++t)
#pragma unroll
    for (int r = 0; r < 8; ++r) {
      size_t idx = (size_t)(m0 + r + 8 * half) * ldc + n0 + t * 16 + l;
      C[idx] = R[idx] + acc[t][r];
    }
}

// ---------------------------------------------------------------------------
// FFN stage 1 (TDM version): T = silu(A@Wg^T) * (A@Wu^T)
// Workgroup = 8 waves, each owns a 16-row M tile; the 64-column weight panels
// for Wg and Wu (64 x K bf16 each) are DMA'd into LDS once by the TDM.
// ---------------------------------------------------------------------------
__global__ void k_gateup_tdm(const bf16_t* __restrict__ A, const bf16_t* __restrict__ Wg,
                             const bf16_t* __restrict__ Wu, bf16_t* __restrict__ T,
                             int K) {
  using namespace cfg;
  extern __shared__ __align__(16) bf16_t smem[];
  bf16_t* sWg = smem;
  bf16_t* sWu = smem + 64 * K;
  const int lane = threadIdx.x & 31, l = lane & 15, half = lane >> 4;
  const int wave = threadIdx.x >> 5;
  const int m0 = blockIdx.x * 128 + wave * 16;
  const int n0p = blockIdx.y * 64;

#if HAVE_TDM
  if (wave == 0) {
    tdm_load_2d(0u, Wg + (size_t)n0p * K, (unsigned)K, 64u, (unsigned)K);
    tdm_load_2d((unsigned)(64 * K * 2), Wu + (size_t)n0p * K, (unsigned)K, 64u, (unsigned)K);
    __builtin_amdgcn_s_wait_tensorcnt(0);
  }
#else
  for (int i = threadIdx.x; i < 64 * K; i += 256) {
    sWg[i] = Wg[(size_t)n0p * K + i];
    sWu[i] = Wu[(size_t)n0p * K + i];
  }
#endif
  __syncthreads();

  v8f ag[4] = {zero8(), zero8(), zero8(), zero8()};
  v8f au[4] = {zero8(), zero8(), zero8(), zero8()};
  for (int k0 = 0; k0 < K; k0 += 32) {
    v16bf a = load_a(A, K, m0 + l, k0, lane);
    __builtin_prefetch(A + (size_t)(m0 + l) * K + k0 + 64, 0, 1);
#pragma unroll
    for (int t = 0; t < 4; ++t) {
      ag[t] = wmma_bf16(a, load_b(sWg, K, t * 16 + l, k0, lane), ag[t]);
      au[t] = wmma_bf16(a, load_b(sWu, K, t * 16 + l, k0, lane), au[t]);
    }
  }
#pragma unroll
  for (int t = 0; t < 4; ++t)
#pragma unroll
    for (int r = 0; r < 8; ++r) {
      float g = ag[t][r];
      float v = (g / (1.f + __expf(-g))) * au[t][r];
      T[(size_t)(m0 + r + 8 * half) * FF + n0p + t * 16 + l] = (bf16_t)v;
    }
}

// ---------------------------------------------------------------------------
// FFN stage 2 (TDM version): O += scale(row) * (T@Wd^T); scale = route[row,e] or 1
// ---------------------------------------------------------------------------
__global__ void k_down_tdm(const bf16_t* __restrict__ T, const bf16_t* __restrict__ Wd,
                           float* __restrict__ O, const float* __restrict__ route,
                           int expert, int K) {
  using namespace cfg;
  extern __shared__ __align__(16) bf16_t smem[];
  bf16_t* sWd = smem;
  const int lane = threadIdx.x & 31, l = lane & 15, half = lane >> 4;
  const int wave = threadIdx.x >> 5;
  const int m0 = blockIdx.x * 128 + wave * 16;
  const int n0p = blockIdx.y * 64;

#if HAVE_TDM
  if (wave == 0) {
    tdm_load_2d(0u, Wd + (size_t)n0p * K, (unsigned)K, 64u, (unsigned)K);
    __builtin_amdgcn_s_wait_tensorcnt(0);
  }
#else
  for (int i = threadIdx.x; i < 64 * K; i += 256)
    sWd[i] = Wd[(size_t)n0p * K + i];
#endif
  __syncthreads();

  v8f acc[4] = {zero8(), zero8(), zero8(), zero8()};
  for (int k0 = 0; k0 < K; k0 += 32) {
    v16bf a = load_a(T, K, m0 + l, k0, lane);
    __builtin_prefetch(T + (size_t)(m0 + l) * K + k0 + 64, 0, 1);
#pragma unroll
    for (int t = 0; t < 4; ++t)
      acc[t] = wmma_bf16(a, load_b(sWd, K, t * 16 + l, k0, lane), acc[t]);
  }
#pragma unroll
  for (int t = 0; t < 4; ++t)
#pragma unroll
    for (int r = 0; r < 8; ++r) {
      int row = m0 + r + 8 * half;
      float sc = route ? route[(size_t)row * NE + expert] : 1.f;
      O[(size_t)row * D + n0p + t * 16 + l] += sc * acc[t][r];
    }
}

// ---------------------------------------------------------------------------
// RoPE in-place on q,k (bf16 [M, D] token-major, head offset h*HD)
// ---------------------------------------------------------------------------
__global__ void k_rope(bf16_t* __restrict__ Q, bf16_t* __restrict__ K) {
  using namespace cfg;
  int idx = blockIdx.x * blockDim.x + threadIdx.x;
  if (idx >= B * S * H * (HD / 2)) return;
  int i = idx & 31; int rest = idx >> 5;
  int h = rest % H; rest /= H;
  int s = rest % S; int b = rest / S;
  size_t base = ((size_t)(b * S + s)) * D + h * HD;
  float inv = __expf(-(float)(2 * i) * (1.0f / (float)HD) * 9.21034037198f);
  float ang = (float)s * inv, sn, cs;
  __sincosf(ang, &sn, &cs);
  float q0 = (float)Q[base + i], q1 = (float)Q[base + 32 + i];
  Q[base + i]      = (bf16_t)(q0 * cs - q1 * sn);
  Q[base + 32 + i] = (bf16_t)(q1 * cs + q0 * sn);
  float k0 = (float)K[base + i], k1 = (float)K[base + 32 + i];
  K[base + i]      = (bf16_t)(k0 * cs - k1 * sn);
  K[base + 32 + i] = (bf16_t)(k1 * cs + k0 * sn);
}

// V [M, D] -> VT [B][H][HD][S]
__global__ void k_vtrans(const bf16_t* __restrict__ V, bf16_t* __restrict__ VT) {
  using namespace cfg;
  int idx = blockIdx.x * blockDim.x + threadIdx.x;
  if (idx >= B * H * HD * S) return;
  int s = idx & (S - 1); int rest = idx >> 10;
  int d = rest & (HD - 1); rest >>= 6;
  int h = rest % H; int b = rest / H;
  VT[idx] = V[((size_t)(b * S + s)) * D + h * HD + d];
}

// ---------------------------------------------------------------------------
// Flash attention: one wave per (16 query rows, head, batch).
// QK^T and P*V via WMMA; online softmax with half-wave shuffles; P staged in LDS.
// ---------------------------------------------------------------------------
__global__ void k_attn(const bf16_t* __restrict__ Qr, const bf16_t* __restrict__ Kr,
                       const bf16_t* __restrict__ VT, bf16_t* __restrict__ O) {
  using namespace cfg;
  const int lane = threadIdx.x & 31, l = lane & 15, half = lane >> 4;
  const int qb = blockIdx.x, h = blockIdx.y, b = blockIdx.z;

  const bf16_t* qbase = Qr + ((size_t)b * S) * D + h * HD;
  const bf16_t* kbase = Kr + ((size_t)b * S) * D + h * HD;
  const bf16_t* vbase = VT + ((size_t)(b * H + h) * HD) * S;

  __shared__ __align__(16) bf16_t pT[16][32];

  v16bf aq0 = load_a(qbase, D, qb * 16 + l, 0, lane);
  v16bf aq1 = load_a(qbase, D, qb * 16 + l, 32, lane);

  v8f oacc[4] = {zero8(), zero8(), zero8(), zero8()};
  float mrow[8], lrow[8];
#pragma unroll
  for (int r = 0; r < 8; ++r) { mrow[r] = -1e30f; lrow[r] = 0.f; }

  const int nkb = (qb * 16 + 16 + 31) >> 5;  // causal: key blocks of 32
  for (int kb = 0; kb < nkb; ++kb) {
    v8f s0 = zero8(), s1 = zero8();
    s0 = wmma_bf16(aq0, load_b(kbase, D, kb * 32 + l,      0,  lane), s0);
    s1 = wmma_bf16(aq0, load_b(kbase, D, kb * 32 + 16 + l, 0,  lane), s1);
    s0 = wmma_bf16(aq1, load_b(kbase, D, kb * 32 + l,      32, lane), s0);
    s1 = wmma_bf16(aq1, load_b(kbase, D, kb * 32 + 16 + l, 32, lane), s1);

#pragma unroll
    for (int r = 0; r < 8; ++r) {
      int qrow = qb * 16 + r + 8 * half;
      int c0 = kb * 32 + l, c1 = c0 + 16;
      float v0 = (c0 <= qrow) ? s0[r] * 0.125f : -1e30f;  // 1/sqrt(64)
      float v1 = (c1 <= qrow) ? s1[r] * 0.125f : -1e30f;
      float mx   = hmax16(fmaxf(v0, v1));
      float mnew = fmaxf(mrow[r], mx);
      float sc   = __expf(mrow[r] - mnew);
      float p0   = __expf(v0 - mnew);
      float p1   = __expf(v1 - mnew);
      lrow[r] = lrow[r] * sc + hsum16(p0 + p1);
      mrow[r] = mnew;
#pragma unroll
      for (int d = 0; d < 4; ++d) oacc[d][r] *= sc;
      s0[r] = p0; s1[r] = p1;
    }

    __syncthreads();
#pragma unroll
    for (int r = 0; r < 8; ++r) {
      pT[r + 8 * half][l]      = (bf16_t)s0[r];
      pT[r + 8 * half][l + 16] = (bf16_t)s1[r];
    }
    __syncthreads();
    v16bf pa = load_a(&pT[0][0], 32, l, 0, lane);

#pragma unroll
    for (int d = 0; d < 4; ++d)
      oacc[d] = wmma_bf16(pa, load_b(vbase, S, d * 16 + l, kb * 32, lane), oacc[d]);
  }

  bf16_t* obase = O + ((size_t)(b * S + qb * 16)) * D + h * HD;
#pragma unroll
  for (int d = 0; d < 4; ++d)
#pragma unroll
    for (int r = 0; r < 8; ++r)
      obase[(size_t)(r + 8 * half) * D + d * 16 + l] = (bf16_t)(oacc[d][r] / lrow[r]);
}

// ---------------------------------------------------------------------------
// Router: sigmoid(h@Wr^T + rb), top-2, normalized weights (zeros elsewhere)
// ---------------------------------------------------------------------------
__global__ void k_router(const bf16_t* __restrict__ Hm, const float* __restrict__ Wr,
                         const float* __restrict__ rb, float* __restrict__ Wout) {
  using namespace cfg;
  int t = blockIdx.x, lane = threadIdx.x & 31;
  float p = -1.f;
  if (lane < NE) {
    const bf16_t* hrow = Hm + (size_t)t * D;
    const float* w = Wr + (size_t)lane * D;
    float acc = 0.f;
    for (int k = 0; k < D; ++k) acc += (float)hrow[k] * w[k];
    acc += rb[lane];
    p = 1.f / (1.f + __expf(-acc));
  }
  float probs[NE];
#pragma unroll
  for (int e = 0; e < NE; ++e) probs[e] = __shfl(p, e, 32);
  int i1 = 0; float p1 = probs[0];
#pragma unroll
  for (int e = 1; e < NE; ++e) if (probs[e] > p1) { p1 = probs[e]; i1 = e; }
  int i2 = -1; float p2 = -2.f;
#pragma unroll
  for (int e = 0; e < NE; ++e) if (e != i1 && probs[e] > p2) { p2 = probs[e]; i2 = e; }
  float denom = p1 + p2;
  if (lane < NE) {
    float w = (lane == i1) ? p1 / denom : ((lane == i2) ? p2 / denom : 0.f);
    Wout[(size_t)t * NE + lane] = w;
  }
}

// ---------------------------------------------------------------------------
// Host-side orchestration
// ---------------------------------------------------------------------------
extern "C" void kernel_launch(void* const* d_in, const int* in_sizes, int n_in,
                              void* d_out, int out_size, void* d_ws, size_t ws_size,
                              hipStream_t stream) {
  using namespace cfg;
  (void)in_sizes; (void)n_in; (void)out_size; (void)ws_size;

  const float* x    = (const float*)d_in[0];
  const float* ln1  = (const float*)d_in[1];
  const float* ln2  = (const float*)d_in[2];
  const float* Wqd  = (const float*)d_in[3];
  const float* Wkvd = (const float*)d_in[4];
  const float* Wqu  = (const float*)d_in[5];
  const float* Wku  = (const float*)d_in[6];
  const float* Wvu  = (const float*)d_in[7];
  const float* Wo   = (const float*)d_in[8];
  const float* sg   = (const float*)d_in[9];
  const float* su   = (const float*)d_in[10];
  const float* sd   = (const float*)d_in[11];
  const float* rg   = (const float*)d_in[12];
  const float* ru   = (const float*)d_in[13];
  const float* rd   = (const float*)d_in[14];
  const float* Wr   = (const float*)d_in[15];
  const float* rb   = (const float*)d_in[16];
  float* out = (float*)d_out;

  char* p = (char*)d_ws;
  auto alloc = [&](size_t bytes) -> char* {
    char* r = p; p += (bytes + 255) & ~(size_t)255; return r;
  };

  bf16_t* h1    = (bf16_t*)alloc((size_t)M * D * 2);
  bf16_t* h2    = (bf16_t*)alloc((size_t)M * D * 2);
  bf16_t* qlat  = (bf16_t*)alloc((size_t)M * LATP * 2);
  bf16_t* kvlat = (bf16_t*)alloc((size_t)M * LATP * 2);
  bf16_t* qb    = (bf16_t*)alloc((size_t)M * D * 2);
  bf16_t* kb    = (bf16_t*)alloc((size_t)M * D * 2);
  bf16_t* vb    = (bf16_t*)alloc((size_t)M * D * 2);
  bf16_t* vt    = (bf16_t*)alloc((size_t)B * H * HD * S * 2);
  bf16_t* ob    = (bf16_t*)alloc((size_t)M * D * 2);
  bf16_t* tbuf  = (bf16_t*)alloc((size_t)M * FF * 2);
  float*  x2    = (float*)alloc((size_t)M * D * 4);
  float*  route = (float*)alloc((size_t)M * NE * 4);
  bf16_t* wqd   = (bf16_t*)alloc((size_t)LAT * D * 2);
  bf16_t* wkvd  = (bf16_t*)alloc((size_t)LAT * D * 2);
  bf16_t* wqu   = (bf16_t*)alloc((size_t)D * LATP * 2);
  bf16_t* wku   = (bf16_t*)alloc((size_t)D * LATP * 2);
  bf16_t* wvu   = (bf16_t*)alloc((size_t)D * LATP * 2);
  bf16_t* wwo   = (bf16_t*)alloc((size_t)D * D * 2);
  bf16_t* wsg   = (bf16_t*)alloc((size_t)FF * D * 2);
  bf16_t* wsu   = (bf16_t*)alloc((size_t)FF * D * 2);
  bf16_t* wsd   = (bf16_t*)alloc((size_t)D * FF * 2);
  bf16_t* wrg   = (bf16_t*)alloc((size_t)NE * FF * D * 2);
  bf16_t* wru   = (bf16_t*)alloc((size_t)NE * FF * D * 2);
  bf16_t* wrd   = (bf16_t*)alloc((size_t)NE * D * FF * 2);

  auto cvt = [&](const float* src, bf16_t* dst, int rows, int cols, int dld) {
    int total = rows * dld;
    k_cvt<<<(total + 255) / 256, 256, 0, stream>>>(src, dst, rows, cols, dld);
  };

  // ---- weight conversions (bf16, padded where needed) ----
  cvt(Wqd,  wqd,  LAT, D, D);
  cvt(Wkvd, wkvd, LAT, D, D);
  cvt(Wqu,  wqu,  D, LAT, LATP);
  cvt(Wku,  wku,  D, LAT, LATP);
  cvt(Wvu,  wvu,  D, LAT, LATP);
  cvt(Wo,   wwo,  D, D, D);
  cvt(sg,   wsg,  FF, D, D);
  cvt(su,   wsu,  FF, D, D);
  cvt(sd,   wsd,  D, FF, FF);
  cvt(rg,   wrg,  NE * FF, D, D);
  cvt(ru,   wru,  NE * FF, D, D);
  cvt(rd,   wrd,  NE * D, FF, FF);

  // zero latent buffers so padded K columns (144..159) are exactly 0
  hipMemsetAsync(qlat,  0, (size_t)M * LATP * 2, stream);
  hipMemsetAsync(kvlat, 0, (size_t)M * LATP * 2, stream);

  // ---- attention path ----
  k_rmsnorm<<<M, 256, 0, stream>>>(x, ln1, h1, D);

  dim3 w32(32);
  k_gemm_bf16out<<<dim3(M / 16, LAT / 16), w32, 0, stream>>>(h1, wqd,  qlat,  D, D, LATP, D);
  k_gemm_bf16out<<<dim3(M / 16, LAT / 16), w32, 0, stream>>>(h1, wkvd, kvlat, D, D, LATP, D);

  k_gemm4_bf16out<<<dim3(M / 16, D / 64), w32, 0, stream>>>(qlat,  wqu, qb, LATP, LATP, D, LATP);
  k_gemm4_bf16out<<<dim3(M / 16, D / 64), w32, 0, stream>>>(kvlat, wku, kb, LATP, LATP, D, LATP);
  k_gemm4_bf16out<<<dim3(M / 16, D / 64), w32, 0, stream>>>(kvlat, wvu, vb, LATP, LATP, D, LATP);

  {
    int total = B * S * H * (HD / 2);
    k_rope<<<(total + 255) / 256, 256, 0, stream>>>(qb, kb);
  }
  {
    int total = B * H * HD * S;
    k_vtrans<<<(total + 255) / 256, 256, 0, stream>>>(vb, vt);
  }

  k_attn<<<dim3(S / 16, H, B), w32, 0, stream>>>(qb, kb, vt, ob);

  // x2 = x + o @ Wo^T
  k_gemm4_f32res<<<dim3(M / 16, D / 64), w32, 0, stream>>>(ob, wwo, x, x2, D, D, D, D);

  // d_out starts as the post-attention residual
  hipMemcpyAsync(out, x2, (size_t)M * D * sizeof(float), hipMemcpyDeviceToDevice, stream);

  // ---- MoE path ----
  k_rmsnorm<<<M, 256, 0, stream>>>(x2, ln2, h2, D);
  k_router<<<M, w32, 0, stream>>>(h2, Wr, rb, route);

  const size_t lds_gu = (size_t)2 * 64 * D * sizeof(bf16_t);   // Wg+Wu panels
  const size_t lds_dn = (size_t)64 * FF * sizeof(bf16_t);      // Wd panel

  // shared expert
  k_gateup_tdm<<<dim3(M / 128, FF / 64), 256, lds_gu, stream>>>(h2, wsg, wsu, tbuf, D);
  k_down_tdm<<<dim3(M / 128, D / 64), 256, lds_dn, stream>>>(tbuf, wsd, out, nullptr, 0, FF);

  // routed experts (dense over all 7; router weight zeros the non-selected)
  for (int e = 0; e < NE; ++e) {
    const bf16_t* eg = wrg + (size_t)e * FF * D;
    const bf16_t* eu = wru + (size_t)e * FF * D;
    const bf16_t* ed = wrd + (size_t)e * D * FF;
    k_gateup_tdm<<<dim3(M / 128, FF / 64), 256, lds_gu, stream>>>(h2, eg, eu, tbuf, D);
    k_down_tdm<<<dim3(M / 128, D / 64), 256, lds_dn, stream>>>(tbuf, ed, out, route, e, FF);
  }
}
